// EGNNEncoder_11261404250494
// MI455X (gfx1250) — matrix operand order repaired and checked by reading, not compiled, not run on hardware
//
#include <hip/hip_runtime.h>
#include <hip/hip_bf16.h>

#define NN 32768
#define NE 262144
#define HD 128
#define NL 4

typedef __attribute__((ext_vector_type(16))) _Float16 v16h;
typedef __attribute__((ext_vector_type(8)))  _Float16 v8h;
typedef __attribute__((ext_vector_type(8)))  float    v8f;

union F16Frag { v16h v; v8h h[2]; };

// A fragment (16x32 f16) from LDS row-major [*, 128]:
// lane L: row = mbase + (L&15); k0 = kk + 8*(L>=16); halves [k0..k0+7],[k0+16..k0+23]
__device__ __forceinline__ v16h load_a_lds(const _Float16* __restrict__ sA, int lane, int mbase, int kk) {
    const int row = mbase + (lane & 15);
    const int k0  = kk + ((lane >> 4) << 3);
    F16Frag f;
    f.h[0] = *(const v8h*)(sA + row * HD + k0);
    f.h[1] = *(const v8h*)(sA + row * HD + k0 + 16);
    return f.v;
}

// B fragment (32x16 f16) from global transposed weight wT[n][k] (row-major, K=128):
// lane L: col = nn + (L&15); k0 = kk + 8*(L>=16); halves [k0..k0+7],[k0+16..k0+23]
__device__ __forceinline__ v16h load_b_gl(const _Float16* __restrict__ wT, int lane, int nn, int kk) {
    const int col = nn + (lane & 15);
    const int k0  = kk + ((lane >> 4) << 3);
    const _Float16* p = wT + col * HD + k0;
    F16Frag f;
    f.h[0] = *(const v8h*)(p);
    f.h[1] = *(const v8h*)(p + 16);
    return f.v;
}

__device__ __forceinline__ float silu_f(float x) { return x / (1.0f + __expf(-x)); }

#define WMMA_F16(a, b, c) \
    __builtin_amdgcn_wmma_f32_16x16x32_f16(false, (a), false, (b), (short)0, (c), false, false)

// ---- CDNA5 async global->LDS copy (ASYNCcnt path); 128 bytes per call -------------
__device__ __forceinline__ void async_row_copy128(unsigned int ldsaddr, unsigned long long gaddr) {
#define A16(O) asm volatile("global_load_async_to_lds_b128 %0, %1, off offset:" #O \
                            :: "v"(ldsaddr), "v"(gaddr) : "memory")
    A16(0); A16(16); A16(32); A16(48); A16(64); A16(80); A16(96); A16(112);
#undef A16
}

__device__ __forceinline__ void wait_async0() {
    asm volatile("s_wait_asynccnt 0x0" ::: "memory");
}

// ---------------- weight pack: f32 [128][128] -> f16 transposed [n][k] ----------------
__global__ void pack128(const float* __restrict__ src, _Float16* __restrict__ dst) {
    int idx = blockIdx.x * 256 + threadIdx.x;   // 64 blocks x 256 = 16384
    int n = idx >> 7, k = idx & 127;
    dst[idx] = (_Float16)src[k * HD + n];
}

// ---------------- utility kernels ----------------
__global__ void zero_f32(float* __restrict__ p, size_t n) {
    size_t i  = (size_t)blockIdx.x * blockDim.x + threadIdx.x;
    size_t st = (size_t)gridDim.x * blockDim.x;
    for (; i < n; i += st) p[i] = 0.0f;
}

__global__ void pos_add(float* __restrict__ pos, const float* __restrict__ del) {
    int i = blockIdx.x * 256 + threadIdx.x;
    if (i < NN * 3) pos[i] += del[i];
}

__global__ void copy_out(const float* __restrict__ hbuf, const float* __restrict__ posbuf,
                         float* __restrict__ out) {
    size_t i  = (size_t)blockIdx.x * 256 + threadIdx.x;
    size_t nh = (size_t)NN * HD;
    if (i < nh) out[i] = hbuf[i];
    else if (i < nh + (size_t)NN * 3) out[i] = posbuf[i - nh];
}

// ---------------- projection: h = x @ proj_w + b ----------------
__global__ __launch_bounds__(256) void proj_kernel(
    const float* __restrict__ x, const _Float16* __restrict__ projT,
    const float* __restrict__ pb, float* __restrict__ hbuf, _Float16* __restrict__ h16) {
    __shared__ _Float16 sX[128 * HD];
    __shared__ float s_b[HD];
    const int tid = threadIdx.x, lane = tid & 31, wv = tid >> 5;
    const int n0 = blockIdx.x * 128;
    if (tid < HD) s_b[tid] = pb[tid];
    {   // stage x tile, convert f32 -> f16
        int r = tid >> 1, off = (tid & 1) * 64;
        const float* src = x + (size_t)(n0 + r) * HD + off;
        _Float16* dst = sX + r * HD + off;
        #pragma unroll
        for (int j = 0; j < 64; ++j) dst[j] = (_Float16)src[j];
    }
    __syncthreads();
    const int mrow = wv * 16, lhi = (lane >> 4) * 8, ln = lane & 15;
    v8f acc[8];
    #pragma unroll
    for (int nt = 0; nt < 8; ++nt) {
        float b = s_b[nt * 16 + ln];
        #pragma unroll
        for (int i = 0; i < 8; ++i) acc[nt][i] = b;
    }
    for (int kt = 0; kt < 4; ++kt) {
        v16h a = load_a_lds(sX, lane, mrow, kt * 32);
        #pragma unroll
        for (int nt = 0; nt < 8; ++nt) {
            v16h b = load_b_gl(projT, lane, nt * 16, kt * 32);
            acc[nt] = WMMA_F16(a, b, acc[nt]);
        }
    }
    #pragma unroll
    for (int nt = 0; nt < 8; ++nt)
        #pragma unroll
        for (int i = 0; i < 8; ++i) {
            size_t idx = (size_t)(n0 + mrow + lhi + i) * HD + nt * 16 + ln;
            float v = acc[nt][i];
            hbuf[idx] = v;
            h16[idx]  = (_Float16)v;
        }
}

// ------- stage gathered h rows into LDS via async global->LDS (no VGPR round-trip) -----
__device__ __forceinline__ void stage_gather_async(_Float16* sH,
                                                   const _Float16* h16,
                                                   const int* ids, int tid) {
    int r = tid >> 1, off = (tid & 1) * 64;   // 128 bytes per thread
    unsigned long long src = (unsigned long long)(size_t)(h16 + (size_t)ids[r] * HD + off);
    unsigned int dst = (unsigned int)(size_t)(sH + r * HD + off);
    async_row_copy128(dst, src);
}

// ---------------- edge kernel: messages + coord weights + scatter ----------------
__global__ __launch_bounds__(256) void edge_kernel(
    const _Float16* __restrict__ h16, const float* __restrict__ pos,
    const int* __restrict__ erow, const int* __restrict__ ecol,
    const _Float16* __restrict__ e1aT, const _Float16* __restrict__ e1bT,
    const float* __restrict__ e1c, const float* __restrict__ eb1,
    const _Float16* __restrict__ e2T, const float* __restrict__ eb2,
    const _Float16* __restrict__ c1T, const float* __restrict__ cb1,
    const float* __restrict__ c2,
    float* __restrict__ agg, float* __restrict__ posdel) {
    __shared__ _Float16 sH[128 * HD];           // reused for h_row / h_col / m1 / m / t
    __shared__ float sdist[128], sinv[128];
    __shared__ float sdiff[128 * 3];
    __shared__ int   srow[128], scol[128];
    __shared__ float s_eb1[HD], s_e1c[HD], s_eb2[HD], s_cb1[HD], s_c2[HD];

    const int tid = threadIdx.x, lane = tid & 31, wv = tid >> 5;
    const int e0 = blockIdx.x * 128;

    // warm L2/WGP$ with the weight matrices while we stage (global_prefetch_b8)
    __builtin_prefetch((const char*)e1aT + tid * 128, 0, 1);
    __builtin_prefetch((const char*)e1bT + tid * 128, 0, 1);
    __builtin_prefetch((const char*)e2T  + tid * 128, 0, 1);
    __builtin_prefetch((const char*)c1T  + tid * 128, 0, 1);

    if (tid < 128) {
        int e = e0 + tid;
        int r = erow[e], c = ecol[e];
        srow[tid] = r; scol[tid] = c;
        float dx = pos[r * 3 + 0] - pos[c * 3 + 0];
        float dy = pos[r * 3 + 1] - pos[c * 3 + 1];
        float dz = pos[r * 3 + 2] - pos[c * 3 + 2];
        float d = sqrtf(dx * dx + dy * dy + dz * dz);
        d = fmaxf(d, 1e-6f);
        sdist[tid] = d;
        sinv[tid]  = 1.0f / d;
        sdiff[tid * 3 + 0] = dx; sdiff[tid * 3 + 1] = dy; sdiff[tid * 3 + 2] = dz;
    } else {
        int j = tid - 128;
        s_eb1[j] = eb1[j]; s_e1c[j] = e1c[j]; s_eb2[j] = eb2[j];
        s_cb1[j] = cb1[j]; s_c2[j] = c2[j];
    }
    __syncthreads();

    stage_gather_async(sH, h16, srow, tid);     // A = h[row], async -> LDS
    wait_async0();
    __syncthreads();

    const int mrow = wv * 16, lhi = (lane >> 4) * 8, ln = lane & 15;
    v8f acc[8];
    // init: bias + dist * w1c  (the 2H+1 concat's last column folded in)
    #pragma unroll
    for (int nt = 0; nt < 8; ++nt) {
        int n = nt * 16 + ln;
        float b = s_eb1[n], wc = s_e1c[n];
        #pragma unroll
        for (int i = 0; i < 8; ++i) acc[nt][i] = b + sdist[mrow + lhi + i] * wc;
    }
    // pass A: h[row] @ W1[0:128]
    for (int kt = 0; kt < 4; ++kt) {
        v16h a = load_a_lds(sH, lane, mrow, kt * 32);
        #pragma unroll
        for (int nt = 0; nt < 8; ++nt) {
            v16h b = load_b_gl(e1aT, lane, nt * 16, kt * 32);
            acc[nt] = WMMA_F16(a, b, acc[nt]);
        }
    }
    __syncthreads();
    stage_gather_async(sH, h16, scol, tid);     // A = h[col], async -> LDS
    wait_async0();
    __syncthreads();
    // pass B: h[col] @ W1[128:256]
    for (int kt = 0; kt < 4; ++kt) {
        v16h a = load_a_lds(sH, lane, mrow, kt * 32);
        #pragma unroll
        for (int nt = 0; nt < 8; ++nt) {
            v16h b = load_b_gl(e1bT, lane, nt * 16, kt * 32);
            acc[nt] = WMMA_F16(a, b, acc[nt]);
        }
    }
    // silu -> m1 written back to this wave's own rows (wave-private, no barrier)
    #pragma unroll
    for (int nt = 0; nt < 8; ++nt)
        #pragma unroll
        for (int i = 0; i < 8; ++i)
            sH[(mrow + lhi + i) * HD + nt * 16 + ln] = (_Float16)silu_f(acc[nt][i]);

    // GEMM2: m = silu(m1 @ W2 + b2); scatter agg atomics
    #pragma unroll
    for (int nt = 0; nt < 8; ++nt) {
        float b = s_eb2[nt * 16 + ln];
        #pragma unroll
        for (int i = 0; i < 8; ++i) acc[nt][i] = b;
    }
    for (int kt = 0; kt < 4; ++kt) {
        v16h a = load_a_lds(sH, lane, mrow, kt * 32);
        #pragma unroll
        for (int nt = 0; nt < 8; ++nt) {
            v16h b = load_b_gl(e2T, lane, nt * 16, kt * 32);
            acc[nt] = WMMA_F16(a, b, acc[nt]);
        }
    }
    #pragma unroll
    for (int nt = 0; nt < 8; ++nt)
        #pragma unroll
        for (int i = 0; i < 8; ++i) {
            int er = mrow + lhi + i, n = nt * 16 + ln;
            float v = silu_f(acc[nt][i]);
            atomicAdd(&agg[(size_t)srow[er] * HD + n], v);
            sH[er * HD + n] = (_Float16)v;      // m, for coord MLP
        }

    // GEMM3: t = silu(m @ coord_w1 + cb1)
    #pragma unroll
    for (int nt = 0; nt < 8; ++nt) {
        float b = s_cb1[nt * 16 + ln];
        #pragma unroll
        for (int i = 0; i < 8; ++i) acc[nt][i] = b;
    }
    for (int kt = 0; kt < 4; ++kt) {
        v16h a = load_a_lds(sH, lane, mrow, kt * 32);
        #pragma unroll
        for (int nt = 0; nt < 8; ++nt) {
            v16h b = load_b_gl(c1T, lane, nt * 16, kt * 32);
            acc[nt] = WMMA_F16(a, b, acc[nt]);
        }
    }
    #pragma unroll
    for (int nt = 0; nt < 8; ++nt)
        #pragma unroll
        for (int i = 0; i < 8; ++i)
            sH[(mrow + lhi + i) * HD + nt * 16 + ln] = (_Float16)silu_f(acc[nt][i]);
    __syncthreads();

    // cw = clip(t . coord_w2, -1, 1); scatter pos delta
    {
        int e = tid >> 1, half = tid & 1;
        const _Float16* tp = sH + e * HD + half * 64;
        const float* cp = s_c2 + half * 64;
        float sum = 0.0f;
        #pragma unroll
        for (int k = 0; k < 64; ++k) sum += (float)tp[k] * cp[k];
        sum += __shfl_xor(sum, 1, 32);
        if (half == 0) {
            float cw = fminf(fmaxf(sum, -1.0f), 1.0f);
            float sc = cw * sinv[e];
            int node = srow[e];
            atomicAdd(&posdel[node * 3 + 0], sdiff[e * 3 + 0] * sc);
            atomicAdd(&posdel[node * 3 + 1], sdiff[e * 3 + 1] * sc);
            atomicAdd(&posdel[node * 3 + 2], sdiff[e * 3 + 2] * sc);
        }
    }
}

// ---------------- node update: h += silu(concat(h,agg)@W1+b1)@W2+b2 ----------------
__global__ __launch_bounds__(256) void node_kernel(
    float* __restrict__ hbuf, _Float16* __restrict__ h16, const float* __restrict__ agg,
    const _Float16* __restrict__ n1aT, const _Float16* __restrict__ n1bT,
    const _Float16* __restrict__ n2T,
    const float* __restrict__ nb1, const float* __restrict__ nb2) {
    __shared__ _Float16 sX[128 * HD];
    __shared__ float s_b1[HD], s_b2[HD];
    const int tid = threadIdx.x, lane = tid & 31, wv = tid >> 5;
    const int n0 = blockIdx.x * 128;
    if (tid < HD) { s_b1[tid] = nb1[tid]; s_b2[tid] = nb2[tid]; }
    {   // stage h rows (already f16): async global -> LDS
        int r = tid >> 1, off = (tid & 1) * 64;
        unsigned long long src = (unsigned long long)(size_t)(h16 + (size_t)(n0 + r) * HD + off);
        unsigned int dst = (unsigned int)(size_t)(sX + r * HD + off);
        async_row_copy128(dst, src);
        wait_async0();
    }
    __syncthreads();
    const int mrow = wv * 16, lhi = (lane >> 4) * 8, ln = lane & 15;
    v8f acc[8];
    #pragma unroll
    for (int nt = 0; nt < 8; ++nt) {
        float b = s_b1[nt * 16 + ln];
        #pragma unroll
        for (int i = 0; i < 8; ++i) acc[nt][i] = b;
    }
    for (int kt = 0; kt < 4; ++kt) {
        v16h a = load_a_lds(sX, lane, mrow, kt * 32);
        #pragma unroll
        for (int nt = 0; nt < 8; ++nt) {
            v16h b = load_b_gl(n1aT, lane, nt * 16, kt * 32);
            acc[nt] = WMMA_F16(a, b, acc[nt]);
        }
    }
    __syncthreads();
    {   // restage agg rows, convert f32 -> f16
        int r = tid >> 1, off = (tid & 1) * 64;
        const float* src = agg + (size_t)(n0 + r) * HD + off;
        _Float16* dst = sX + r * HD + off;
        #pragma unroll
        for (int j = 0; j < 64; ++j) dst[j] = (_Float16)src[j];
    }
    __syncthreads();
    for (int kt = 0; kt < 4; ++kt) {
        v16h a = load_a_lds(sX, lane, mrow, kt * 32);
        #pragma unroll
        for (int nt = 0; nt < 8; ++nt) {
            v16h b = load_b_gl(n1bT, lane, nt * 16, kt * 32);
            acc[nt] = WMMA_F16(a, b, acc[nt]);
        }
    }
    #pragma unroll
    for (int nt = 0; nt < 8; ++nt)
        #pragma unroll
        for (int i = 0; i < 8; ++i)
            sX[(mrow + lhi + i) * HD + nt * 16 + ln] = (_Float16)silu_f(acc[nt][i]);

    #pragma unroll
    for (int nt = 0; nt < 8; ++nt) {
        float b = s_b2[nt * 16 + ln];
        #pragma unroll
        for (int i = 0; i < 8; ++i) acc[nt][i] = b;
    }
    for (int kt = 0; kt < 4; ++kt) {
        v16h a = load_a_lds(sX, lane, mrow, kt * 32);
        #pragma unroll
        for (int nt = 0; nt < 8; ++nt) {
            v16h b = load_b_gl(n2T, lane, nt * 16, kt * 32);
            acc[nt] = WMMA_F16(a, b, acc[nt]);
        }
    }
    #pragma unroll
    for (int nt = 0; nt < 8; ++nt)
        #pragma unroll
        for (int i = 0; i < 8; ++i) {
            size_t idx = (size_t)(n0 + mrow + lhi + i) * HD + nt * 16 + ln;
            float hn = hbuf[idx] + acc[nt][i];
            hbuf[idx] = hn;
            h16[idx]  = (_Float16)hn;
        }
}

// ---------------- host launcher ----------------
extern "C" void kernel_launch(void* const* d_in, const int* in_sizes, int n_in,
                              void* d_out, int out_size, void* d_ws, size_t ws_size,
                              hipStream_t stream) {
    (void)in_sizes; (void)n_in; (void)out_size; (void)ws_size;
    const float* x        = (const float*)d_in[0];
    const float* pos_in   = (const float*)d_in[1];
    const int*   eidx     = (const int*)d_in[2];
    const float* proj_w   = (const float*)d_in[3];
    const float* proj_b   = (const float*)d_in[4];
    const float* edge_w1  = (const float*)d_in[5];
    const float* edge_b1  = (const float*)d_in[6];
    const float* edge_w2  = (const float*)d_in[7];
    const float* edge_b2  = (const float*)d_in[8];
    const float* node_w1  = (const float*)d_in[9];
    const float* node_b1  = (const float*)d_in[10];
    const float* node_w2  = (const float*)d_in[11];
    const float* node_b2  = (const float*)d_in[12];
    const float* coord_w1 = (const float*)d_in[13];
    const float* coord_b1 = (const float*)d_in[14];
    const float* coord_w2 = (const float*)d_in[15];

    char* ws = (char*)d_ws;
    size_t off = 0;
    auto take = [&](size_t bytes) -> char* {
        char* p = ws + off;
        off += (bytes + 255) & ~(size_t)255;
        return p;
    };
    float*    hbuf   = (float*)take((size_t)NN * HD * 4);
    float*    agg    = (float*)take((size_t)NN * HD * 4);
    _Float16* h16    = (_Float16*)take((size_t)NN * HD * 2);
    float*    posbuf = (float*)take((size_t)NN * 3 * 4);
    float*    posdel = (float*)take((size_t)NN * 3 * 4);
    _Float16* projT  = (_Float16*)take((size_t)HD * HD * 2);
    _Float16* e1aT   = (_Float16*)take((size_t)NL * HD * HD * 2);
    _Float16* e1bT   = (_Float16*)take((size_t)NL * HD * HD * 2);
    _Float16* e2T    = (_Float16*)take((size_t)NL * HD * HD * 2);
    _Float16* c1T    = (_Float16*)take((size_t)NL * HD * HD * 2);
    _Float16* n1aT   = (_Float16*)take((size_t)NL * HD * HD * 2);
    _Float16* n1bT   = (_Float16*)take((size_t)NL * HD * HD * 2);
    _Float16* n2T    = (_Float16*)take((size_t)NL * HD * HD * 2);

    // pack weights (f32 -> f16, transposed to [n][k])
    pack128<<<64, 256, 0, stream>>>(proj_w, projT);
    for (int l = 0; l < NL; ++l) {
        pack128<<<64, 256, 0, stream>>>(edge_w1 + (size_t)l * 257 * HD,            e1aT + (size_t)l * HD * HD);
        pack128<<<64, 256, 0, stream>>>(edge_w1 + (size_t)l * 257 * HD + 128 * HD, e1bT + (size_t)l * HD * HD);
        pack128<<<64, 256, 0, stream>>>(edge_w2 + (size_t)l * HD * HD,             e2T  + (size_t)l * HD * HD);
        pack128<<<64, 256, 0, stream>>>(coord_w1 + (size_t)l * HD * HD,            c1T  + (size_t)l * HD * HD);
        pack128<<<64, 256, 0, stream>>>(node_w1 + (size_t)l * 256 * HD,            n1aT + (size_t)l * HD * HD);
        pack128<<<64, 256, 0, stream>>>(node_w1 + (size_t)l * 256 * HD + 128 * HD, n1bT + (size_t)l * HD * HD);
        pack128<<<64, 256, 0, stream>>>(node_w2 + (size_t)l * HD * HD,             n2T  + (size_t)l * HD * HD);
    }
    (void)hipMemcpyAsync(posbuf, pos_in, (size_t)NN * 3 * 4, hipMemcpyDeviceToDevice, stream);

    proj_kernel<<<NN / 128, 256, 0, stream>>>(x, projT, proj_b, hbuf, h16);

    for (int l = 0; l < NL; ++l) {
        zero_f32<<<1024, 256, 0, stream>>>(agg, (size_t)NN * HD);
        zero_f32<<<256, 256, 0, stream>>>(posdel, (size_t)NN * 3);
        edge_kernel<<<NE / 128, 256, 0, stream>>>(
            h16, posbuf, eidx, eidx + NE,
            e1aT + (size_t)l * HD * HD, e1bT + (size_t)l * HD * HD,
            edge_w1 + (size_t)l * 257 * HD + 256 * HD,   // w1c row
            edge_b1 + (size_t)l * HD,
            e2T + (size_t)l * HD * HD, edge_b2 + (size_t)l * HD,
            c1T + (size_t)l * HD * HD, coord_b1 + (size_t)l * HD,
            coord_w2 + (size_t)l * HD,
            agg, posdel);
        node_kernel<<<NN / 128, 256, 0, stream>>>(
            hbuf, h16, agg,
            n1aT + (size_t)l * HD * HD, n1bT + (size_t)l * HD * HD,
            n2T + (size_t)l * HD * HD,
            node_b1 + (size_t)l * HD, node_b2 + (size_t)l * HD);
        pos_add<<<(NN * 3 + 255) / 256, 256, 0, stream>>>(posbuf, posdel);
    }

    size_t tot = (size_t)NN * HD + (size_t)NN * 3;
    copy_out<<<(unsigned)((tot + 255) / 256), 256, 0, stream>>>(hbuf, posbuf, (float*)d_out);
}
